// HexSlidingAttentionLayer_18648747999878
// MI455X (gfx1250) — matrix operand order repaired and checked by reading, not compiled, not run on hardware
//
#include <hip/hip_runtime.h>
#include <stdint.h>

// ---------------------------------------------------------------------------
// HexSlidingAttentionLayer on MI455X (gfx1250, wave32, WMMA bf16 + TDM
// double-buffered B-tile staging)
// ---------------------------------------------------------------------------
#define D_MODEL   256
#define NHEAD     8
#define HEAD_DIM  32
#define DFF       1024
#define BB        16
#define HH        64
#define WW        64
#define NTOK      (BB * HH * WW)      // 65536 tokens
#define EPSL      1e-5f

typedef __attribute__((ext_vector_type(16))) __bf16 v16bf;
typedef __attribute__((ext_vector_type(8)))  float  v8f;
typedef unsigned int v4u_t __attribute__((ext_vector_type(4)));
typedef int          v8i_t __attribute__((ext_vector_type(8)));
typedef int          v4i_t __attribute__((ext_vector_type(4)));

#if defined(__has_builtin)
#if __has_builtin(__builtin_amdgcn_tensor_load_to_lds) && \
    __has_builtin(__builtin_amdgcn_s_wait_tensorcnt)
#define USE_TDM 1
#endif
#endif
#ifndef USE_TDM
#define USE_TDM 0
#endif

union Frag {            // 16 bf16 = 32 bytes = 8 VGPRs
    v16bf v;
    uint4 u[2];
};

__device__ __forceinline__ uint16_t f2bf(float f) {
    uint32_t u = __float_as_uint(f);
    uint32_t r = (u + 0x7FFFu + ((u >> 16) & 1u)) >> 16;   // round-to-nearest-even
    return (uint16_t)r;
}
__device__ __forceinline__ float bf2f(uint16_t x) {
    return __uint_as_float(((uint32_t)x) << 16);
}

#if USE_TDM
// Issue a TDM load of a 64x64-element (2B) B chunk into LDS.
// D# pad fields reproduce the padded 144-byte LDS row stride:
// pad_interval=4 -> pad after 32 DWORDs (128B row), pad_amount=3 -> +4 DWORDs (16B).
__device__ __forceinline__ void tdm_load_b_tile(const uint16_t* gptr,
                                                uint32_t ldsAddr, int K) {
    const uint64_t ga = (uint64_t)(uintptr_t)gptr;
    v4u_t g0;
    g0.x = 1u;                                     // count=1 (valid), load, no gather
    g0.y = ldsAddr;                                // lds_addr
    g0.z = (uint32_t)ga;                           // global_addr[31:0]
    g0.w = (uint32_t)((ga >> 32) & 0x01FFFFFFu)    // global_addr[56:32]
         | (2u << 30);                             // type = 2 ("image")
    v8i_t g1;
    g1[0] = (int)((1u << 16)      // data_size = 1 -> 2 bytes
                | (1u << 20)      // pad_enable
                | (4u << 22)      // pad_interval: pad after 32 DWORDs (128B)
                | (3u << 25));    // pad_amount: 4 DWORDs (16B) -> 144B LDS stride
    g1[1] = (int)(((uint32_t)K & 0xFFFFu) << 16);          // tensor_dim0[15:0]
    g1[2] = (int)((((uint32_t)K >> 16) & 0xFFFFu)          // tensor_dim0[31:16]
                | (0xFFFFu << 16));                        // tensor_dim1[15:0]=65535
    g1[3] = (int)(64u << 16);                              // tile_dim0 = 64
    g1[4] = 64;                                            // tile_dim1 = 64
    g1[5] = K;                                             // tensor_dim0_stride
    g1[6] = 0;
    g1[7] = 0;
    v4i_t z4 = {0, 0, 0, 0};
#if __clang_major__ >= 23
    v8i_t z8 = {0, 0, 0, 0, 0, 0, 0, 0};
    __builtin_amdgcn_tensor_load_to_lds(g0, g1, z4, z4, z8, 0);
#else
    __builtin_amdgcn_tensor_load_to_lds(g0, g1, z4, z4, 0);
#endif
}
#endif

// ---------------------------------------------------------------------------
// f32 -> bf16 elementwise convert
// ---------------------------------------------------------------------------
__global__ __launch_bounds__(256) void cvt_f32_bf16(const float* __restrict__ in,
                                                    uint16_t* __restrict__ out, int n) {
    int i = blockIdx.x * 256 + threadIdx.x;
    if (i < n) out[i] = f2bf(in[i]);
}

// ---------------------------------------------------------------------------
// Tiled bf16 WMMA GEMM:  C[M,N] = A[M,K] * B[N,K]^T  (+ bias, opt. ReLU)
//   A row-major bf16 (lda = K), B row-major bf16 [N][K], output row-major ldc
//   Workgroup tile: 128(M) x 64(N), 8 waves, each wave 16(M) x 64(N)
//   B tiles DMA'd by the Tensor Data Mover into ping-pong LDS buffers so the
//   transfer of chunk k+1 overlaps the WMMAs of chunk k (TENSORcnt is
//   in-order per wave, so s_wait_tensorcnt(1) proves chunk k arrived).
//   MODE 0: f32 out (+bias) | MODE 1: bf16 out relu(+bias) | MODE 2: bf16 out (+bias)
// ---------------------------------------------------------------------------
template <int MODE>
__global__ __launch_bounds__(256) void wmma_gemm(const uint16_t* __restrict__ A,
                                                 const uint16_t* __restrict__ B,
                                                 const float* __restrict__ bias,
                                                 void* __restrict__ Cout,
                                                 int K, int ldc) {
    __shared__ uint16_t Bs[2][64][72];   // 2 x (64 cols x 64-K), 144B padded rows

    const int wave = threadIdx.x >> 5;
    const int lane = threadIdx.x & 31;
    const int l16  = lane & 15;
    const bool hi  = lane >= 16;

    const int rowBase = blockIdx.x * 128 + wave * 16;
    const int colBase = blockIdx.y * 64;

    v8f acc[4] = {};

    const uint16_t* __restrict__ Arow  = A + (size_t)(rowBase + l16) * K;
    const uint16_t* __restrict__ Bbase = B + (size_t)colBase * K;

#if USE_TDM
    if (threadIdx.x < 32)   // one wave issues; TDM ignores EXEC, tracked on TENSORcnt
        tdm_load_b_tile(Bbase, (uint32_t)(uintptr_t)&Bs[0][0][0], K);
#endif

    for (int kk = 0; kk < K; kk += 64) {
        const int cur = (kk >> 6) & 1;
#if USE_TDM
        if (threadIdx.x < 32) {
            if (kk + 64 < K) {
                // prefetch next chunk into the alternate buffer, then wait for
                // the current chunk only (in-order completion => cnt<=1 ok)
                tdm_load_b_tile(Bbase + kk + 64,
                                (uint32_t)(uintptr_t)&Bs[cur ^ 1][0][0], K);
                __builtin_amdgcn_s_wait_tensorcnt(1);
            } else {
                __builtin_amdgcn_s_wait_tensorcnt(0);
            }
        }
#else
        // ---- fallback: cooperative stage of B chunk into LDS ----
        {
            const int c     = threadIdx.x >> 2;          // 0..63
            const int kpart = (threadIdx.x & 3) * 16;    // 0,16,32,48
            const uint16_t* srcp = Bbase + (size_t)c * K + kk + kpart;
            uint4 d0 = *(const uint4*)(srcp);
            uint4 d1 = *(const uint4*)(srcp + 8);
            *(uint4*)&Bs[cur][c][kpart]     = d0;
            *(uint4*)&Bs[cur][c][kpart + 8] = d1;
        }
#endif
        if (kk + 64 < K) __builtin_prefetch(Arow + kk + 64, 0, 1);  // global_prefetch_b8
        __syncthreads();

        // ---- batch all fragment loads, then back-to-back WMMAs ----
        Frag a[2], bfr[8];
        const uint16_t* ap = Arow + kk + (hi ? 8 : 0);
        a[0].u[0] = *(const uint4*)(ap);          // K {0..7}/{8..15}
        a[0].u[1] = *(const uint4*)(ap + 16);     // K {16..23}/{24..31}
        a[1].u[0] = *(const uint4*)(ap + 32);     // K {32..39}/{40..47}
        a[1].u[1] = *(const uint4*)(ap + 48);     // K {48..55}/{56..63}
#pragma unroll
        for (int t = 0; t < 2; ++t)
#pragma unroll
            for (int j = 0; j < 4; ++j) {
                const uint16_t* bp = &Bs[cur][j * 16 + l16][t * 32 + (hi ? 8 : 0)];
                bfr[t * 4 + j].u[0] = *(const uint4*)(bp);
                bfr[t * 4 + j].u[1] = *(const uint4*)(bp + 16);
            }
#pragma unroll
        for (int t = 0; t < 2; ++t)
#pragma unroll
            for (int j = 0; j < 4; ++j)
                acc[j] = __builtin_amdgcn_wmma_f32_16x16x32_bf16(
                    false, a[t].v, false, bfr[t * 4 + j].v, (short)0, acc[j], false, false);
        __syncthreads();
    }

    // ---- epilogue: D layout lane<16 -> rows 0..7, lane>=16 -> rows 8..15 ----
    const int row0 = rowBase + (hi ? 8 : 0);
#pragma unroll
    for (int j = 0; j < 4; ++j) {
        const int col = colBase + j * 16 + l16;
        const float bv = bias[col];
#pragma unroll
        for (int r = 0; r < 8; ++r) {
            float v = acc[j][r] + bv;
            size_t off = (size_t)(row0 + r) * ldc + col;
            if (MODE == 0) {
                ((float*)Cout)[off] = v;
            } else if (MODE == 1) {
                v = v > 0.f ? v : 0.f;
                ((uint16_t*)Cout)[off] = f2bf(v);
            } else {
                ((uint16_t*)Cout)[off] = f2bf(v);
            }
        }
    }
}

// ---------------------------------------------------------------------------
// Hex sliding-window attention. One block per token; wave w == head w,
// lane d == head dim. QKV is bf16 [token][768] (Q|K|V), biases already added.
// OOB neighbors contribute K=bk, V=bv (zero feature through projection) and
// still participate in the softmax, matching the reference exactly.
// ---------------------------------------------------------------------------
__global__ __launch_bounds__(256) void hex_attn(const uint16_t* __restrict__ QKV,
                                                const float* __restrict__ in_proj_b,
                                                uint16_t* __restrict__ ctx) {
    const int n   = blockIdx.x;
    const int h   = threadIdx.x >> 5;
    const int d   = threadIdx.x & 31;
    const int ch  = h * HEAD_DIM + d;
    const int b   = n >> 12;          // / (64*64)
    const int rem = n & 4095;
    const int r   = rem >> 6;
    const int c   = rem & 63;

    const float q = bf2f(QKV[(size_t)n * 768 + ch]);

    const int dq[7] = {0, 1, -1, 0, 0, 1, -1};
    const int dr[7] = {0, 0, 0, 1, -1, -1, 1};

    float s[7], vv[7];
    float smax = -3.0e38f;
#pragma unroll
    for (int i = 0; i < 7; ++i) {
        const int nc = c + dq[i];
        const int nr = r + dr[i];
        float kval, vval;
        if (nc >= 0 && nc < WW && nr >= 0 && nr < HH) {
            const size_t nn = ((size_t)b * 4096 + nr * 64 + nc) * 768;
            kval = bf2f(QKV[nn + 256 + ch]);
            vval = bf2f(QKV[nn + 512 + ch]);
        } else {
            kval = in_proj_b[256 + ch];
            vval = in_proj_b[512 + ch];
        }
        vv[i] = vval;
        float p = q * kval;                    // reduce over the head (one wave)
        p += __shfl_xor(p, 16);
        p += __shfl_xor(p, 8);
        p += __shfl_xor(p, 4);
        p += __shfl_xor(p, 2);
        p += __shfl_xor(p, 1);
        s[i] = p * 0.17677669529663687f;       // 1/sqrt(32)
        smax = fmaxf(smax, s[i]);
    }
    float Z = 0.f, num = 0.f;
#pragma unroll
    for (int i = 0; i < 7; ++i) {
        const float e = expf(s[i] - smax);
        Z   += e;
        num += e * vv[i];
    }
    ctx[(size_t)n * D_MODEL + ch] = f2bf(num / Z);
}

// ---------------------------------------------------------------------------
// Residual add + LayerNorm over C=256. One block (256 thr) per token.
// Writes f32 (and optional bf16 copy for the next bf16 GEMM).
// ---------------------------------------------------------------------------
__global__ __launch_bounds__(256) void add_layernorm(const float* __restrict__ base,
                                                     const float* __restrict__ add,
                                                     const float* __restrict__ g,
                                                     const float* __restrict__ beta,
                                                     float* __restrict__ outf,
                                                     uint16_t* __restrict__ outb) {
    __shared__ float red[16];
    const int n = blockIdx.x;
    const int t = threadIdx.x;
    const size_t idx = (size_t)n * D_MODEL + t;

    const float v = base[idx] + add[idx];
    float s = v, s2 = v * v;
    s  += __shfl_xor(s, 16);  s2 += __shfl_xor(s2, 16);
    s  += __shfl_xor(s, 8);   s2 += __shfl_xor(s2, 8);
    s  += __shfl_xor(s, 4);   s2 += __shfl_xor(s2, 4);
    s  += __shfl_xor(s, 2);   s2 += __shfl_xor(s2, 2);
    s  += __shfl_xor(s, 1);   s2 += __shfl_xor(s2, 1);
    const int w = t >> 5;
    if ((t & 31) == 0) { red[w] = s; red[8 + w] = s2; }
    __syncthreads();
    float S = 0.f, S2 = 0.f;
#pragma unroll
    for (int i = 0; i < 8; ++i) { S += red[i]; S2 += red[8 + i]; }

    const float mu  = S * (1.f / D_MODEL);
    const float var = S2 * (1.f / D_MODEL) - mu * mu;
    const float o   = (v - mu) * rsqrtf(var + EPSL) * g[t] + beta[t];
    outf[idx] = o;
    if (outb) outb[idx] = f2bf(o);
}

// ---------------------------------------------------------------------------
// Launch sequence
// ---------------------------------------------------------------------------
extern "C" void kernel_launch(void* const* d_in, const int* in_sizes, int n_in,
                              void* d_out, int out_size, void* d_ws, size_t ws_size,
                              hipStream_t stream) {
    const float* src       = (const float*)d_in[0];
    const float* in_proj_w = (const float*)d_in[1];   // (768,256)
    const float* in_proj_b = (const float*)d_in[2];   // (768)
    const float* out_w     = (const float*)d_in[3];   // (256,256)
    const float* out_b     = (const float*)d_in[4];
    const float* w1        = (const float*)d_in[5];   // (1024,256)
    const float* b1        = (const float*)d_in[6];
    const float* w2        = (const float*)d_in[7];   // (256,1024)
    const float* b2        = (const float*)d_in[8];
    const float* g1        = (const float*)d_in[9];
    const float* beta1     = (const float*)d_in[10];
    const float* g2        = (const float*)d_in[11];
    const float* beta2     = (const float*)d_in[12];
    float* out = (float*)d_out;

    // ---- workspace layout (lifetime-aliased, ~290 MB) ----
    char* ws = (char*)d_ws;
    const size_t MB = 1ull << 20;
    uint16_t* Xb    = (uint16_t*)(ws + 0);           // 32MB  X bf16      (dead after GEMM1)
    uint16_t* x1b   = Xb;                            // 32MB  x1 bf16     (written at LN1)
    uint16_t* QKVb  = (uint16_t*)(ws + 32 * MB);     // 96MB  QKV bf16    (dead after attn)
    uint16_t* Hb    = QKVb;                          // 128MB FFN hidden bf16 [32,160)
    float*    YF    = (float*)(ws + 160 * MB);       // 64MB  attn-out / ffn-out f32
    uint16_t* ctxb  = (uint16_t*)(ws + 224 * MB);    // 32MB  ctx bf16    (dead after GEMM2)
    float*    x1f   = (float*)(ws + 224 * MB);       // 64MB  x1 f32      (written at LN1)
    uint16_t* wqkvb = (uint16_t*)(ws + 288 * MB);    // 384KB
    uint16_t* outwb = wqkvb + 768 * 256;             // 128KB
    uint16_t* w1b   = outwb + 256 * 256;             // 512KB
    uint16_t* w2b   = w1b + 1024 * 256;              // 512KB

    const int NX = NTOK * D_MODEL;                   // 16.7M

    // 1) convert activations + weights to bf16
    cvt_f32_bf16<<<(NX + 255) / 256, 256, 0, stream>>>(src, Xb, NX);
    cvt_f32_bf16<<<(768 * 256 + 255) / 256, 256, 0, stream>>>(in_proj_w, wqkvb, 768 * 256);
    cvt_f32_bf16<<<(256 * 256 + 255) / 256, 256, 0, stream>>>(out_w, outwb, 256 * 256);
    cvt_f32_bf16<<<(1024 * 256 + 255) / 256, 256, 0, stream>>>(w1, w1b, 1024 * 256);
    cvt_f32_bf16<<<(256 * 1024 + 255) / 256, 256, 0, stream>>>(w2, w2b, 256 * 1024);

    // 2) QKV projection: [N,256] x [768,256]^T -> bf16 [N,768] (+in_proj_b)
    wmma_gemm<2><<<dim3(NTOK / 128, 768 / 64), 256, 0, stream>>>(
        Xb, wqkvb, in_proj_b, (void*)QKVb, 256, 768);

    // 3) hex attention -> ctx bf16 [N,256]
    hex_attn<<<NTOK, 256, 0, stream>>>(QKVb, in_proj_b, ctxb);

    // 4) out projection: ctx x out_w^T + out_b -> f32 Y
    wmma_gemm<0><<<dim3(NTOK / 128, 256 / 64), 256, 0, stream>>>(
        ctxb, outwb, out_b, (void*)YF, 256, 256);

    // 5) x1 = LN(src + Y) -> f32 + bf16
    add_layernorm<<<NTOK, 256, 0, stream>>>(src, YF, g1, beta1, x1f, x1b);

    // 6) FFN1: relu(x1 x w1^T + b1) -> bf16 H [N,1024]
    wmma_gemm<1><<<dim3(NTOK / 128, 1024 / 64), 256, 0, stream>>>(
        x1b, w1b, b1, (void*)Hb, 256, 1024);

    // 7) FFN2: H x w2^T + b2 -> f32 F
    wmma_gemm<0><<<dim3(NTOK / 128, 256 / 64), 256, 0, stream>>>(
        Hb, w2b, b2, (void*)YF, 1024, 256);

    // 8) out = LN(x1 + F)
    add_layernorm<<<NTOK, 256, 0, stream>>>(x1f, YF, g2, beta2, out, nullptr);
}
